// BiLSTMCRF_1219770712826
// MI455X (gfx1250) — matrix operand order repaired and checked by reading, not compiled, not run on hardware
//
#include <hip/hip_runtime.h>
#include <hip/hip_bf16.h>

// ---------------------------------------------------------------------------
// BiLSTM-CRF forward loss for MI455X (gfx1250, wave32, WMMA + TDM).
// B=64 T=512 V=50000 E=256 H=256 L=9.  All GEMMs in bf16 WMMA w/ f32 accum.
// ---------------------------------------------------------------------------

#define Bv   64
#define Tv   512
#define Ev   256
#define Hv   256
#define HGv  1024            // 4*H gate width
#define Lv   9
#define Mv   (Bv * Tv)       // 32768 tokens

typedef __bf16 bf16_t;
typedef __attribute__((ext_vector_type(16))) __bf16 v16bf;
typedef __attribute__((ext_vector_type(8)))  __bf16 v8bf;
typedef __attribute__((ext_vector_type(8)))  float  v8f;
typedef __attribute__((ext_vector_type(4)))  unsigned int v4u;
typedef __attribute__((ext_vector_type(8)))  unsigned int v8u;

// ---- bf16 <-> f32 helpers (bit-level, RNE) --------------------------------
static __device__ __forceinline__ float bf2f(bf16_t b) {
    unsigned short u = __builtin_bit_cast(unsigned short, b);
    unsigned int v = ((unsigned int)u) << 16;
    return __builtin_bit_cast(float, v);
}
static __device__ __forceinline__ bf16_t f2bf(float f) {
    unsigned int v = __builtin_bit_cast(unsigned int, f);
    unsigned int lsb = (v >> 16) & 1u;
    v += 0x7FFFu + lsb;                       // round-to-nearest-even
    unsigned short u = (unsigned short)(v >> 16);
    return __builtin_bit_cast(bf16_t, u);
}
static __device__ __forceinline__ v8f vzero8() {
    v8f z;
    #pragma unroll
    for (int i = 0; i < 8; ++i) z[i] = 0.0f;
    return z;
}
static __device__ __forceinline__ float sigmoidf_(float x) {
    return 1.0f / (1.0f + __expf(-x));
}

// ---- WMMA fragment loaders (contiguous-k order, same in A and B) ---------------------------
// A (16x32 bf16, M rows): lane l holds row (l&15); K run = b0..b0+15 contiguous,
// b0 = (l>=16)*16  -> two contiguous 16B loads.
static __device__ __forceinline__ v16bf load_frag_a(const bf16_t* __restrict__ base,
                                                    int ld, int row0, int k0, int lane) {
    int r  = row0 + (lane & 15);
    int kb = k0 + ((lane >> 4) << 4);
    const bf16_t* p = base + (size_t)r * ld + kb;
    union { v16bf v; v8bf h[2]; } u;
    u.h[0] = *(const v8bf*)(p);
    u.h[1] = *(const v8bf*)(p + 8);
    return u.v;
}
// B (32x16 bf16, N cols): weight stored row-major (N,K); column n of B == row n
// of weight.  lane l holds col (l&15), K run = k0 + (l>=16)*16 .. contiguous 16
// -> one 32B load.
static __device__ __forceinline__ v16bf load_frag_b(const bf16_t* __restrict__ base,
                                                    int ld, int n0, int k0, int lane) {
    int r  = n0 + (lane & 15);
    int kb = k0 + ((lane >> 4) << 4);
    const bf16_t* p = base + (size_t)r * ld + kb;
    return *(const v16bf*)(p);
}

#define WMMA_BF16(a, b, c) __builtin_amdgcn_wmma_f32_16x16x32_bf16( \
    false, (a), false, (b), (short)0, (c), false, false)

// ---- Tensor Data Mover: 2D tile (rows x cols bf16) global -> LDS ----------
// D# per ISA cdna5 §8.3/8.4: group0 {count=1,lds_addr,global_addr,type=2},
// group1 {data_size=2B, tensor/tile dims, dim0 stride}.  2-group form.
static __device__ __forceinline__ void tdm_load_2d_to_lds(
        const void* gsrc, unsigned lds_byte_off,
        unsigned rows, unsigned cols_elems, unsigned row_stride_elems) {
    unsigned long long ga = (unsigned long long)gsrc;
    v4u g0;
    g0[0] = 1u;                                               // count=1 (user D#)
    g0[1] = lds_byte_off;                                     // lds_addr
    g0[2] = (unsigned)(ga & 0xFFFFFFFFu);                     // global_addr[31:0]
    g0[3] = (unsigned)((ga >> 32) & 0x01FFFFFFu) | (2u << 30);// ga[56:32] | type=2
    v8u g1;
    g1[0] = (1u << 16);                                       // data_size=1 -> 2B
    g1[1] = (cols_elems & 0xFFFFu) << 16;                     // tensor_dim0 lo
    g1[2] = ((cols_elems >> 16) & 0xFFFFu)                    // tensor_dim0 hi
          | ((rows & 0xFFFFu) << 16);                         // tensor_dim1 lo
    g1[3] = ((rows >> 16) & 0xFFFFu)                          // tensor_dim1 hi
          | ((cols_elems & 0xFFFFu) << 16);                   // tile_dim0
    g1[4] = (rows & 0xFFFFu);                                 // tile_dim1 (tile_dim2=0)
    g1[5] = row_stride_elems;                                 // tensor_dim0_stride lo
    g1[6] = 0u;                                               // stride hi | dim1_stride lo
    g1[7] = 0u;
    asm volatile("tensor_load_to_lds %0, %1" :: "s"(g0), "s"(g1) : "memory");
}

// ---------------------------------------------------------------------------
// 1) f32 -> bf16 weight conversion
// ---------------------------------------------------------------------------
__global__ void cvt_f32_bf16(const float* __restrict__ in, bf16_t* __restrict__ out, int n) {
    int i = blockIdx.x * blockDim.x + threadIdx.x;
    if (i < n) out[i] = f2bf(in[i]);
}

// ---------------------------------------------------------------------------
// 2) Embedding gather: x_bf16[b,t,e] = bf16(emb[ids[b,t], e])
// ---------------------------------------------------------------------------
__global__ void embed_gather(const int* __restrict__ ids, const float* __restrict__ emb,
                             bf16_t* __restrict__ x) {
    size_t i = (size_t)blockIdx.x * blockDim.x + threadIdx.x;
    if (i < (size_t)Mv * Ev) {
        int tok = ids[i >> 8];                       // E == 256
        x[i] = f2bf(emb[(size_t)tok * Ev + (i & 255)]);
    }
}

// ---------------------------------------------------------------------------
// 3) Input projection GEMM:  C[dir][M,N] = A[M,K] @ W[dir][N,K]^T + bias[dir][N]
//    Weight slice (128 x K) staged to LDS by TDM; A from global with register
//    ping-pong so loads overlap WMMA.  Block = 256 thr (8 waves), tile 128x128,
//    wave = 4 Mtiles x 2 Ntiles.  __launch_bounds__(256,1) -> no VGPR cap/spills.
// ---------------------------------------------------------------------------
static __device__ __forceinline__ void load_a4(v16bf af[4], const bf16_t* __restrict__ A,
                                               int Ksz, int m0, int k0, int lane) {
    #pragma unroll
    for (int i = 0; i < 4; ++i) af[i] = load_frag_a(A, Ksz, m0 + i * 16, k0, lane);
}
static __device__ __forceinline__ void mm8(v8f acc[4][2], const v16bf af[4],
                                           const bf16_t* __restrict__ Ws,
                                           int Ksz, int n0l, int k0, int lane) {
    v16bf b0 = load_frag_b(Ws, Ksz, n0l,      k0, lane);
    v16bf b1 = load_frag_b(Ws, Ksz, n0l + 16, k0, lane);
    #pragma unroll
    for (int i = 0; i < 4; ++i) {
        acc[i][0] = WMMA_BF16(af[i], b0, acc[i][0]);
        acc[i][1] = WMMA_BF16(af[i], b1, acc[i][1]);
    }
}

__global__ void __launch_bounds__(256, 1)
gemm_bias_bf16(const bf16_t* __restrict__ A, const bf16_t* __restrict__ W,
               const float* __restrict__ bias, bf16_t* __restrict__ C,
               int Msz, int Nsz, int Ksz) {
    extern __shared__ char smem[];
    bf16_t* Ws = (bf16_t*)smem;               // [128][Ksz] staged weight slice

    const int tid  = threadIdx.x;
    const int lane = tid & 31;
    const int wv   = tid >> 5;
    const int wm   = wv >> 2;                 // 0..1
    const int wn   = wv & 3;                  // 0..3
    const int dir  = blockIdx.z;

    const bf16_t* Wd = W + (size_t)dir * Nsz * Ksz + (size_t)blockIdx.x * 128 * Ksz;
    const float*  bd = bias + (size_t)dir * Nsz;
    bf16_t*       Cd = C    + (size_t)dir * Msz * Nsz;

    // ---- TDM: stage 128 x K weight tile into LDS (one wave issues) ----
    if (wv == 0) {
        tdm_load_2d_to_lds(Wd, 0, 128u, (unsigned)Ksz, (unsigned)Ksz);
        __builtin_amdgcn_s_wait_tensorcnt(0);
    }
    __syncthreads();

    const int m0  = blockIdx.y * 128 + wm * 64;
    const int n0l = wn * 32;                  // local N base inside staged slice

    v8f acc[4][2];
    #pragma unroll
    for (int i = 0; i < 4; ++i)
        #pragma unroll
        for (int j = 0; j < 2; ++j) acc[i][j] = vzero8();

    const int nk = Ksz >> 5;                  // 32-wide K steps (8 or 16; even)
    v16bf a0[4], a1[4];
    load_a4(a0, A, Ksz, m0, 0, lane);
    int kt = 0;
    for (;;) {
        if (kt + 1 < nk) load_a4(a1, A, Ksz, m0, (kt + 1) * 32, lane);
        mm8(acc, a0, Ws, Ksz, n0l, kt * 32, lane);
        if (++kt == nk) break;
        if (kt + 1 < nk) load_a4(a0, A, Ksz, m0, (kt + 1) * 32, lane);
        mm8(acc, a1, Ws, Ksz, n0l, kt * 32, lane);
        if (++kt == nk) break;
    }

    // C layout (ISA): vgpr r, lane l -> M = m0+r+(l>=16)*8, N = n0+(l&15)
    #pragma unroll
    for (int i = 0; i < 4; ++i) {
        #pragma unroll
        for (int j = 0; j < 2; ++j) {
            int col   = blockIdx.x * 128 + n0l + j * 16 + (lane & 15);
            float bv  = bd[col];
            int rbase = m0 + i * 16 + ((lane >> 4) << 3);
            #pragma unroll
            for (int r = 0; r < 8; ++r)
                Cd[(size_t)(rbase + r) * Nsz + col] = f2bf(acc[i][j][r] + bv);
        }
    }
}

// ---------------------------------------------------------------------------
// 4) Recurrent BiLSTM layer.  grid.x = 2 (dir: 0 fwd, 1 bwd), block = 1024
//    (32 waves -> 8 waves/SIMD, so VGPR cap is tight: keep live state small).
//    Per step: gates(64x1024) = h(64x256) @ Whh[dir]^T + gx[dir][:,t,:],
//    then elementwise cell update.  h/c/gates live in dynamic LDS (224 KB).
//    Wave w -> Mtile (w&3)*16, 8 N-tiles at (w>>2)*128, processed in two
//    halves of 4 tiles so only acc[4]+a+bfr[4] (~75 VGPRs) are live.
// ---------------------------------------------------------------------------
__global__ void __launch_bounds__(1024)
lstm_layer(const bf16_t* __restrict__ gx,   // [2][Mv][HGv]
           const bf16_t* __restrict__ Whh,  // [2][HGv][Hv]
           bf16_t* __restrict__ out) {      // [Bv][Tv][2*Hv]
    extern __shared__ char smem[];
    bf16_t* h_lds = (bf16_t*)smem;                                   // 64*256 bf16
    float*  c_lds = (float*)(smem + Bv * Hv * 2);                    // 64*256 f32
    bf16_t* g_lds = (bf16_t*)(smem + Bv * Hv * 2 + Bv * Hv * 4);     // 64*1024 bf16

    const int d    = blockIdx.x;
    const int tid  = threadIdx.x;
    const int lane = tid & 31;
    const int wv   = tid >> 5;                 // 0..31
    const int mt   = (wv & 3) * 16;            // row tile base (0..48)
    const int nb   = (wv >> 2) * 128;          // col base (8 tiles of 16)

    const bf16_t* Wd  = Whh + (size_t)d * HGv * Hv;
    const bf16_t* gxd = gx  + (size_t)d * Mv * HGv;

    for (int p = tid; p < Bv * Hv; p += 1024) { h_lds[p] = f2bf(0.0f); c_lds[p] = 0.0f; }
    __syncthreads();

    for (int step = 0; step < Tv; ++step) {
        const int t = d ? (Tv - 1 - step) : step;

        // ---- gates = h @ Whh^T (WMMA) + gx, staged to LDS ----
        #pragma unroll
        for (int half = 0; half < 2; ++half) {
            const int nh = nb + half * 64;     // 4 tiles this half
            v8f acc[4];
            #pragma unroll
            for (int j = 0; j < 4; ++j) acc[j] = vzero8();

            #pragma unroll
            for (int kt = 0; kt < 8; ++kt) {
                const int k0 = kt * 32;
                v16bf a = load_frag_a(h_lds, Hv, mt, k0, lane);
                v16bf bfr[4];
                #pragma unroll
                for (int j = 0; j < 4; ++j)
                    bfr[j] = load_frag_b(Wd, Hv, nh + j * 16, k0, lane);
                #pragma unroll
                for (int j = 0; j < 4; ++j)
                    acc[j] = WMMA_BF16(a, bfr[j], acc[j]);
            }

            #pragma unroll
            for (int j = 0; j < 4; ++j) {
                const int col   = nh + j * 16 + (lane & 15);
                const int rbase = mt + ((lane >> 4) << 3);
                #pragma unroll
                for (int r = 0; r < 8; ++r) {
                    const int row = rbase + r;   // batch index
                    float v = acc[j][r] + bf2f(gxd[((size_t)row * Tv + t) * HGv + col]);
                    g_lds[row * HGv + col] = f2bf(v);
                }
            }
        }
        __syncthreads();

        // ---- prefetch next step's gx slab (overlaps elementwise phase) ----
        if (step + 1 < Tv) {
            const int tn = d ? (Tv - 2 - step) : (step + 1);
            const bf16_t* pf = gxd + ((size_t)(tid >> 4) * Tv + tn) * HGv + (tid & 15) * 64;
            __builtin_prefetch(pf, 0, 0);
        }

        // ---- elementwise LSTM cell ----
        for (int p = tid; p < Bv * Hv; p += 1024) {
            const int m = p >> 8, j = p & 255;
            float ig = sigmoidf_(bf2f(g_lds[m * HGv + j]));
            float fg = sigmoidf_(bf2f(g_lds[m * HGv + Hv + j]));
            float gg = tanhf(bf2f(g_lds[m * HGv + 2 * Hv + j]));
            float og = sigmoidf_(bf2f(g_lds[m * HGv + 3 * Hv + j]));
            float c  = fg * c_lds[p] + ig * gg;
            float h  = og * tanhf(c);
            c_lds[p] = c;
            bf16_t hb = f2bf(h);
            h_lds[p] = hb;
            out[((size_t)m * Tv + t) * (2 * Hv) + d * Hv + j] = hb;
        }
        __syncthreads();
    }
}

// ---------------------------------------------------------------------------
// 5) Emission head: em[m,l] = sum_k out1[m,k]*lin_w[l,k] + lin_b[l]  (L=9)
// ---------------------------------------------------------------------------
__global__ void emissions_kernel(const bf16_t* __restrict__ x,   // [Mv][2H]
                                 const float* __restrict__ lw,   // [L][2H]
                                 const float* __restrict__ lb,
                                 float* __restrict__ em) {       // [Mv][L]
    int tid = blockIdx.x * blockDim.x + threadIdx.x;
    if (tid >= Mv * Lv) return;
    int m = tid / Lv, l = tid % Lv;
    const bf16_t* xr = x + (size_t)m * (2 * Hv);
    const float*  wr = lw + (size_t)l * (2 * Hv);
    float s = lb[l];
    #pragma unroll 8
    for (int k = 0; k < 2 * Hv; ++k) s += bf2f(xr[k]) * wr[k];
    em[(size_t)m * Lv + l] = s;
}

// ---------------------------------------------------------------------------
// 6) CRF negative log-likelihood.  One thread per batch element (B=64).
// ---------------------------------------------------------------------------
__global__ void crf_kernel(const float* __restrict__ em,      // [B][T][L]
                           const int* __restrict__ labels,    // [B][T]
                           const int* __restrict__ mask,      // [B][T]
                           const float* __restrict__ trans,   // [L][L]
                           const float* __restrict__ st, const float* __restrict__ et,
                           float* __restrict__ out) {
    __shared__ float red[Bv];
    const int b = threadIdx.x;
    const float* eb = em + (size_t)b * Tv * Lv;
    const int* lb = labels + (size_t)b * Tv;
    const int* mb = mask   + (size_t)b * Tv;

    float alpha[Lv];
    const int l0 = lb[0];
    float score = st[l0] + eb[l0];
    #pragma unroll
    for (int j = 0; j < Lv; ++j) alpha[j] = st[j] + eb[j];

    int cnt = (mb[0] != 0) ? 1 : 0;
    for (int t = 1; t < Tv; ++t) {
        const int mt = mb[t];
        const int lt = lb[t];
        if (mt) {
            score += trans[lb[t - 1] * Lv + lt] + eb[t * Lv + lt];
            ++cnt;
            float na[Lv];
            #pragma unroll
            for (int j = 0; j < Lv; ++j) {
                float mx = -1e30f;
                #pragma unroll
                for (int i = 0; i < Lv; ++i)
                    mx = fmaxf(mx, alpha[i] + trans[i * Lv + j]);
                float s = 0.0f;
                #pragma unroll
                for (int i = 0; i < Lv; ++i)
                    s += __expf(alpha[i] + trans[i * Lv + j] - mx);
                na[j] = mx + __logf(s) + eb[t * Lv + j];
            }
            #pragma unroll
            for (int j = 0; j < Lv; ++j) alpha[j] = na[j];
        }
    }
    score += et[lb[cnt - 1]];

    float mx = -1e30f;
    #pragma unroll
    for (int j = 0; j < Lv; ++j) mx = fmaxf(mx, alpha[j] + et[j]);
    float s = 0.0f;
    #pragma unroll
    for (int j = 0; j < Lv; ++j) s += __expf(alpha[j] + et[j] - mx);
    const float logZ = mx + __logf(s);

    red[b] = -(score - logZ);
    __syncthreads();
    if (b == 0) {
        float acc = 0.0f;
        for (int i = 0; i < Bv; ++i) acc += red[i];
        out[0] = acc / (float)Bv;
    }
}

// ---------------------------------------------------------------------------
// Host-side orchestration
// ---------------------------------------------------------------------------
static inline size_t align256(size_t x) { return (x + 255) & ~(size_t)255; }

extern "C" void kernel_launch(void* const* d_in, const int* in_sizes, int n_in,
                              void* d_out, int out_size, void* d_ws, size_t ws_size,
                              hipStream_t stream) {
    const int*   ids    = (const int*)d_in[0];
    const int*   amask  = (const int*)d_in[1];
    const int*   labels = (const int*)d_in[2];
    const float* emb    = (const float*)d_in[3];
    const float* w_ih0  = (const float*)d_in[4];
    const float* w_hh0  = (const float*)d_in[5];
    const float* b0     = (const float*)d_in[6];
    const float* w_ih1  = (const float*)d_in[7];
    const float* w_hh1  = (const float*)d_in[8];
    const float* b1     = (const float*)d_in[9];
    const float* lin_w  = (const float*)d_in[10];
    const float* lin_b  = (const float*)d_in[11];
    const float* trans  = (const float*)d_in[12];
    const float* st     = (const float*)d_in[13];
    const float* et     = (const float*)d_in[14];
    float* loss_out     = (float*)d_out;

    // ---- workspace layout (bytes) ----
    char* ws = (char*)d_ws;
    size_t off = 0;
    bf16_t* x_bf   = (bf16_t*)(ws + off); off = align256(off + (size_t)Mv * Ev * 2);            // 16 MB
    bf16_t* gx     = (bf16_t*)(ws + off); off = align256(off + (size_t)2 * Mv * HGv * 2);       // 128 MB
    bf16_t* out0   = (bf16_t*)(ws + off); off = align256(off + (size_t)Mv * 2 * Hv * 2);        // 32 MB
    bf16_t* out1   = (bf16_t*)(ws + off); off = align256(off + (size_t)Mv * 2 * Hv * 2);        // 32 MB
    float*  emis   = (float*)(ws + off);  off = align256(off + (size_t)Mv * Lv * 4);            // 1.1 MB
    bf16_t* wih0b  = (bf16_t*)(ws + off); off = align256(off + (size_t)2 * HGv * Ev * 2);
    bf16_t* whh0b  = (bf16_t*)(ws + off); off = align256(off + (size_t)2 * HGv * Hv * 2);
    bf16_t* wih1b  = (bf16_t*)(ws + off); off = align256(off + (size_t)2 * HGv * 2 * Hv * 2);
    bf16_t* whh1b  = (bf16_t*)(ws + off); off = align256(off + (size_t)2 * HGv * Hv * 2);

    // 1) weight conversion to bf16
    {
        int n;
        n = 2 * HGv * Ev;      cvt_f32_bf16<<<(n + 255) / 256, 256, 0, stream>>>(w_ih0, wih0b, n);
        n = 2 * HGv * Hv;      cvt_f32_bf16<<<(n + 255) / 256, 256, 0, stream>>>(w_hh0, whh0b, n);
        n = 2 * HGv * 2 * Hv;  cvt_f32_bf16<<<(n + 255) / 256, 256, 0, stream>>>(w_ih1, wih1b, n);
        n = 2 * HGv * Hv;      cvt_f32_bf16<<<(n + 255) / 256, 256, 0, stream>>>(w_hh1, whh1b, n);
    }

    // 2) embedding gather (f32 -> bf16)
    {
        size_t n = (size_t)Mv * Ev;
        embed_gather<<<(unsigned)((n + 255) / 256), 256, 0, stream>>>(ids, emb, x_bf);
    }

    // 3) layer-0 input projection: gx[dir] = x @ w_ih0[dir]^T + b0[dir]
    gemm_bias_bf16<<<dim3(HGv / 128, Mv / 128, 2), 256, (size_t)128 * Ev * 2, stream>>>(
        x_bf, wih0b, b0, gx, Mv, HGv, Ev);

    // 4) layer-0 recurrence (fwd + bwd concurrently), 224 KB dynamic LDS
    const size_t lds_bytes = (size_t)Bv * Hv * 2 + (size_t)Bv * Hv * 4 + (size_t)Bv * HGv * 2;
    lstm_layer<<<2, 1024, lds_bytes, stream>>>(gx, whh0b, out0);

    // 5) layer-1 input projection: gx[dir] = out0 @ w_ih1[dir]^T + b1[dir]
    gemm_bias_bf16<<<dim3(HGv / 128, Mv / 128, 2), 256, (size_t)128 * 2 * Hv * 2, stream>>>(
        out0, wih1b, b1, gx, Mv, HGv, 2 * Hv);

    // 6) layer-1 recurrence
    lstm_layer<<<2, 1024, lds_bytes, stream>>>(gx, whh1b, out1);

    // 7) emissions
    {
        int n = Mv * Lv;
        emissions_kernel<<<(n + 255) / 256, 256, 0, stream>>>(out1, lin_w, lin_b, emis);
    }

    // 8) CRF NLL -> scalar loss
    crf_kernel<<<1, Bv, 0, stream>>>(emis, labels, amask, trans, st, et, loss_out);
}
